// TaggerBiRNNCRF_12652973654336
// MI455X (gfx1250) — matrix operand order repaired and checked by reading, not compiled, not run on hardware
//
#include <hip/hip_runtime.h>
#include <hip/hip_bf16.h>

// ---------- types for CDNA5 WMMA ----------
typedef __attribute__((ext_vector_type(16))) __bf16 v16bf;
typedef __attribute__((ext_vector_type(8)))  __bf16 v8bf;
typedef __attribute__((ext_vector_type(8)))  float  v8f;
typedef __attribute__((ext_vector_type(8)))  unsigned short u16x8;
typedef __attribute__((ext_vector_type(4)))  unsigned short u16x4;
typedef __attribute__((ext_vector_type(4)))  float  f4v;

#define B_  64
#define S_  512
#define E_  128
#define H_  256
#define G_  1024   // 4*H gates
#define KC  384    // H + E fused K dimension
#define KT  20
#define KP  32     // padded tag dim for WMMA N tiles
#define SOS 19
#define NEGV (-9999.0f)

#define NTILE 64          // N tiles per direction (1024/16)
#define KSTEPS 12         // K steps (384/32)
#define FRAG 16           // bf16 elements per lane per fragment
#define TILE_STRIDE (KSTEPS * 32 * FRAG)   // elements per tile in swizzled W

__device__ __forceinline__ unsigned short f2bf(float f) {
  union { float f; unsigned u; } x; x.f = f;
  unsigned r = x.u + 0x7FFFu + ((x.u >> 16) & 1u);   // round-to-nearest-even
  return (unsigned short)(r >> 16);
}
__device__ __forceinline__ float fast_tanh(float x) {
  x = fminf(15.f, fmaxf(-15.f, x));
  float e = __expf(2.f * x);
  return (e - 1.f) / (e + 1.f);
}
__device__ __forceinline__ float sigmoidf(float x) {
  return 1.f / (1.f + __expf(-x));
}

// ---------- weight prep ----------
// LSTM weights: f32 [W_hh | W_ih] -> bf16 swizzled into exact B-fragment order:
//   Wsw[tile][ks][lane][16], tile=n/16, ks=k/32; lane<16: n=tile*16+lane, k=ks*32+j
//   lane>=16: same n, k=ks*32+16+j  => one wave fragment load = 1KB fully coalesced
__global__ void prep_kernel(const float* __restrict__ W_ih_f, const float* __restrict__ W_hh_f,
                            const float* __restrict__ W_ih_b, const float* __restrict__ W_hh_b,
                            const float* __restrict__ lin_W,
                            unsigned short* __restrict__ WswF,
                            unsigned short* __restrict__ WswB,
                            unsigned short* __restrict__ linWp) {
  int i = blockIdx.x * blockDim.x + threadIdx.x;
  const int nW = NTILE * TILE_STRIDE;        // 393216 per direction
  if (i < 2 * nW) {
    int d = i / nW;
    int r = i - d * nW;
    int j    = r & 15;
    int lane = (r >> 4) & 31;
    int t3   = r >> 9;           // tile*12 + ks
    int ks   = t3 % KSTEPS;
    int tile = t3 / KSTEPS;
    int n = tile * 16 + (lane & 15);
    int k = ks * 32 + (lane >> 4) * 16 + j;
    const float* Whh = d ? W_hh_b : W_hh_f;
    const float* Wih = d ? W_ih_b : W_ih_f;
    float v = (k < H_) ? Whh[n * H_ + k] : Wih[n * E_ + (k - H_)];
    (d ? WswB : WswF)[r] = f2bf(v);
  } else if (i < 2 * nW + KP * 2 * H_) {
    int j = i - 2 * nW;
    int n = j / (2 * H_), d2 = j % (2 * H_);
    linWp[j] = f2bf((n < KT) ? lin_W[n * (2 * H_) + d2] : 0.f);
  }
}

// ---------- fused embed-gather + BiLSTM scan ----------
// grid = 8 blocks: dir = blockIdx>>2, batch quarter = blockIdx&3 (16 rows = 1 WMMA M tile)
// A panel in LDS = [h(16x256) | x(16x128)] contiguous, K = 384 (12 wmma k-steps)
// B streamed from L2-resident swizzled weights; #pragma unroll 2 keeps ks dynamic
// so LICM cannot hoist the loop-invariant loads and spill them to scratch.
__global__ __launch_bounds__(512) void lstm_kernel(
    const int* __restrict__ word_seq, const int* __restrict__ lengths,
    const float* __restrict__ embed,
    const unsigned short* __restrict__ WswF, const unsigned short* __restrict__ WswB,
    const float* __restrict__ b_f, const float* __restrict__ b_b,
    unsigned short* __restrict__ Hf, unsigned short* __restrict__ Hb)
{
  __shared__ __align__(32) unsigned short a_lds[16 * KC];   // 12 KB unified A panel
  __shared__ __align__(32) float          gates[16 * G_];   // 64 KB C tiles
  __shared__ int len_s[16];

  const int dir = blockIdx.x >> 2;
  const int b0  = (blockIdx.x & 3) * 16;
  const unsigned short* W    = dir ? WswB : WswF;
  const float*          bias = dir ? b_b  : b_f;
  unsigned short*       Hout = dir ? Hb   : Hf;

  const int tid  = threadIdx.x;
  const int lane = tid & 31;
  const int wv   = tid >> 5;        // 16 waves
  const int col  = lane & 15;       // tile column / A row
  const int hi   = lane >> 4;       // lane half

  if (tid < 16) len_s[tid] = lengths[b0 + tid];
  for (int i = tid; i < 16 * KC; i += 512) a_lds[i] = 0;
  float c[8];
  #pragma unroll
  for (int q = 0; q < 8; ++q) c[q] = 0.f;

  // bias for this thread's 32 gate lanes, preloaded (removes per-step global loads)
  f4v br[4][2];
  #pragma unroll
  for (int g4 = 0; g4 < 4; ++g4) {
    br[g4][0] = *(const f4v*)(bias + g4 * H_ + lane * 8);
    br[g4][1] = *(const f4v*)(bias + g4 * H_ + lane * 8 + 4);
  }

  // per-lane base into swizzled weights for this wave's 4 tiles
  const unsigned short* bbase = W + (size_t)(wv * 4) * TILE_STRIDE + lane * FRAG;
  __syncthreads();

  const int row_len = len_s[wv];
  const unsigned short* arow = a_lds + col * KC;

  for (int s = 0; s < S_; ++s) {
    // (1) gather+convert x_t: one b128 load + one ds_store_b64 per thread
    {
      const int len = len_s[wv];
      const int t = dir ? ((s < len) ? (len - 1 - s) : s) : s;
      const int w = word_seq[(b0 + wv) * S_ + t];      // wave-uniform
      f4v xv = *(const f4v*)(embed + (size_t)w * E_ + lane * 4);
      u16x4 xb;
      #pragma unroll
      for (int q = 0; q < 4; ++q) xb[q] = f2bf(xv[q]);
      *(u16x4*)(a_lds + wv * KC + H_ + lane * 4) = xb;
    }
    __syncthreads();

    // (2) gate GEMM: A fragment loaded once per k-step, reused by 4 N-tiles
    v8f acc[4];
    #pragma unroll
    for (int nt = 0; nt < 4; ++nt) acc[nt] = (v8f){};
    #pragma unroll 2
    for (int ks = 0; ks < KSTEPS; ++ks) {
      const int kb = ks * 32;
      union { v16bf v; v8bf h[2]; } a;
      a.v = *(const v16bf*)(arow + kb + hi * 16);
      const unsigned short* bks = bbase + ks * (32 * FRAG);
      #pragma unroll
      for (int nt = 0; nt < 4; ++nt) {
        v16bf bfrag = *(const v16bf*)(bks + nt * TILE_STRIDE);
        acc[nt] = __builtin_amdgcn_wmma_f32_16x16x32_bf16(
            false, a.v, false, bfrag, (short)0, acc[nt], false, false);
      }
    }
    #pragma unroll
    for (int nt = 0; nt < 4; ++nt) {
      const int n = wv * 64 + nt * 16 + col;
      #pragma unroll
      for (int v = 0; v < 8; ++v)
        gates[(v + 8 * hi) * G_ + n] = acc[nt][v];
    }
    __syncthreads();

    // (3) pointwise LSTM cell update (wave-uniform mask; 16B vector stores)
    const bool m_ok = (s < row_len);
    u16x8 hb8 = {};
    #pragma unroll
    for (int q = 0; q < 8; ++q) {
      const int n = lane * 8 + q;
      float gi = gates[wv * G_ + n]          + br[0][q >> 2][q & 3];
      float gf = gates[wv * G_ + H_ + n]     + br[1][q >> 2][q & 3];
      float gg = gates[wv * G_ + 2 * H_ + n] + br[2][q >> 2][q & 3];
      float go = gates[wv * G_ + 3 * H_ + n] + br[3][q >> 2][q & 3];
      float ig = sigmoidf(gi), fg = sigmoidf(gf), og = sigmoidf(go);
      float tg = fast_tanh(gg);
      float cn = fg * c[q] + ig * tg;
      float hn = og * fast_tanh(cn);
      if (m_ok) { c[q] = cn; hb8[q] = f2bf(hn); }
    }
    if (m_ok) *(u16x8*)(a_lds + wv * KC + lane * 8) = hb8;   // carry h state
    *(u16x8*)(Hout + ((size_t)(b0 + wv) * S_ + s) * H_ + lane * 8) = hb8;  // hs = h*m
    __syncthreads();
  }
}

// ---------- emission features: feats = [h_f | h_b(ri)] @ lin_W^T, N padded to 32 ----------
__global__ __launch_bounds__(64) void feats_kernel(
    const int* __restrict__ lengths,
    const unsigned short* __restrict__ Hf, const unsigned short* __restrict__ Hb,
    const unsigned short* __restrict__ linWp, const float* __restrict__ lin_b,
    float* __restrict__ featsW)
{
  const int mt   = blockIdx.x;            // M tile over B*S (2048 tiles)
  const int lane = threadIdx.x & 31;
  const int nt   = threadIdx.x >> 5;      // 2 N tiles, one per wave
  const int col  = lane & 15;
  const int hi   = lane >> 4;

  const int r   = mt * 16 + col;          // A row for this lane
  const int b   = r >> 9, s = r & 511;
  const int len = lengths[b];
  const int rib = (s < len) ? (len - 1 - s) : s;      // un-reverse backward states
  const unsigned short* arow_f = Hf + (size_t)r * H_;
  const unsigned short* arow_b = Hb + ((size_t)b * S_ + rib) * H_;
  const unsigned short* wrow   = linWp + (nt * 16 + col) * (2 * H_);

  v8f acc = {};
  #pragma unroll 4
  for (int ks = 0; ks < 16; ++ks) {
    const int kb = ks * 32;
    v16bf bfrag = *(const v16bf*)(wrow + kb + hi * 16);
    union { v16bf v; v8bf h[2]; } a;
    const unsigned short* asrc = (kb < H_) ? (arow_f + kb) : (arow_b + (kb - H_));
    a.v = *(const v16bf*)(asrc + hi * 16);
    acc = __builtin_amdgcn_wmma_f32_16x16x32_bf16(
        false, a.v, false, bfrag, (short)0, acc, false, false);
  }
  const int n  = nt * 16 + col;
  const float bn = (n < KT) ? lin_b[n] : 0.f;
  #pragma unroll
  for (int v = 0; v < 8; ++v) {
    int rr = mt * 16 + v + 8 * hi;
    featsW[(size_t)rr * KP + n] = acc[v] + bn;
  }
}

// ---------- CRF: numerator gather + forward logsumexp recursion ----------
// one wave per batch row; transition row cached in 20 VGPRs per lane
__global__ __launch_bounds__(32) void crf_kernel(
    const int* __restrict__ tags, const int* __restrict__ lengths,
    const float* __restrict__ trans, const float* __restrict__ featsW,
    float* __restrict__ out)
{
  const int b   = blockIdx.x;
  const int j   = threadIdx.x;
  const int len = lengths[b];
  __shared__ float sc[KT];

  float tr[KT];
  if (j < KT) {
    #pragma unroll
    for (int i = 0; i < KT; ++i) tr[i] = trans[j * KT + i];
  }

  // numerator: sum_s mask * (emis + trans[cur, prev])
  float num = 0.f;
  for (int s = j; s < len; s += 32) {
    int tg = tags[b * S_ + s];
    int pv = (s == 0) ? SOS : tags[b * S_ + s - 1];
    num += featsW[((size_t)b * S_ + s) * KP + tg] + trans[tg * KT + pv];
  }
  #pragma unroll
  for (int off = 16; off > 0; off >>= 1) num += __shfl_xor(num, off);

  // forward recursion (only masked steps change score)
  float score = (j == SOS) ? 0.f : NEGV;
  for (int s = 0; s < len; ++s) {
    if (j < KT) sc[j] = score;
    __syncthreads();
    if (j < KT) {
      float em = featsW[((size_t)b * S_ + s) * KP + j];
      float mx = -3.0e38f;
      #pragma unroll
      for (int i = 0; i < KT; ++i) mx = fmaxf(mx, sc[i] + tr[i]);
      float sum = 0.f;
      #pragma unroll
      for (int i = 0; i < KT; ++i) sum += __expf(sc[i] + tr[i] - mx);
      score = em + mx + __logf(sum);
    }
    __syncthreads();
  }
  if (j < KT) sc[j] = score;
  __syncthreads();
  if (j == 0) {
    float mx = -3.0e38f;
    for (int i = 0; i < KT; ++i) mx = fmaxf(mx, sc[i]);
    float sum = 0.f;
    for (int i = 0; i < KT; ++i) sum += __expf(sc[i] - mx);
    out[b] = (mx + __logf(sum)) - num;
  }
}

// ---------- host launcher ----------
extern "C" void kernel_launch(void* const* d_in, const int* in_sizes, int n_in,
                              void* d_out, int out_size, void* d_ws, size_t ws_size,
                              hipStream_t stream) {
  (void)in_sizes; (void)n_in; (void)out_size; (void)ws_size;
  const int*   word_seq = (const int*)  d_in[0];
  const int*   tags     = (const int*)  d_in[1];
  const int*   lengths  = (const int*)  d_in[2];
  const float* embed    = (const float*)d_in[3];
  const float* W_ih_f   = (const float*)d_in[4];
  const float* W_hh_f   = (const float*)d_in[5];
  const float* b_f      = (const float*)d_in[6];
  const float* W_ih_b   = (const float*)d_in[7];
  const float* W_hh_b   = (const float*)d_in[8];
  const float* b_b      = (const float*)d_in[9];
  const float* lin_W    = (const float*)d_in[10];
  const float* lin_b    = (const float*)d_in[11];
  const float* trans    = (const float*)d_in[12];
  float* out = (float*)d_out;

  char* ws = (char*)d_ws;
  size_t off = 0;
  unsigned short* WswF = (unsigned short*)(ws + off); off += (size_t)NTILE * TILE_STRIDE * 2;
  unsigned short* WswB = (unsigned short*)(ws + off); off += (size_t)NTILE * TILE_STRIDE * 2;
  unsigned short* linWp = (unsigned short*)(ws + off); off += (size_t)KP * 2 * H_ * 2;
  off = (off + 255) & ~(size_t)255;
  unsigned short* Hf = (unsigned short*)(ws + off); off += (size_t)B_ * S_ * H_ * 2;
  unsigned short* Hb = (unsigned short*)(ws + off); off += (size_t)B_ * S_ * H_ * 2;
  float* featsW = (float*)(ws + off);               off += (size_t)B_ * S_ * KP * 4;

  const int nprep = 2 * NTILE * TILE_STRIDE + KP * 2 * H_;
  prep_kernel<<<(nprep + 255) / 256, 256, 0, stream>>>(
      W_ih_f, W_hh_f, W_ih_b, W_hh_b, lin_W, WswF, WswB, linWp);
  lstm_kernel<<<8, 512, 0, stream>>>(
      word_seq, lengths, embed, WswF, WswB, b_f, b_b, Hf, Hb);
  feats_kernel<<<(B_ * S_) / 16, 64, 0, stream>>>(
      lengths, Hf, Hb, linWp, lin_b, featsW);
  crf_kernel<<<B_, 32, 0, stream>>>(tags, lengths, trans, featsW, out);
}
